// FCOSDecoder_78116865180102
// MI455X (gfx1250) — compile-verified, matrix operand order checked by reading, not run on hardware
//
#include <hip/hip_runtime.h>
#include <hip/hip_bf16.h>

// ---------------------------------------------------------------------------
// FCOS decoder head for MI455X (gfx1250): implicit-GEMM 3x3 convs via
// v_wmma_f32_16x16x32_bf16, channel-blocked bf16 activations, LDS double
// buffering (branchless zero-pad staging, incremental k-loop bookkeeping),
// BN folded into weights.
// ---------------------------------------------------------------------------

typedef __attribute__((ext_vector_type(16))) __bf16 v16bf;
typedef __attribute__((ext_vector_type(8)))  __bf16 v8bf;
typedef __attribute__((ext_vector_type(8)))  float  v8f;

static __device__ __forceinline__ __bf16 f2bf(float x) {
  unsigned u = __float_as_uint(x);
  unsigned r = (u + 0x7FFFu + ((u >> 16) & 1u)) >> 16;   // round-to-nearest-even
  unsigned short h = (unsigned short)r;
  __bf16 o; __builtin_memcpy(&o, &h, 2); return o;
}
static __device__ __forceinline__ float bf2f(__bf16 b) {
  unsigned short h; __builtin_memcpy(&h, &b, 2);
  return __uint_as_float(((unsigned)h) << 16);
}

// ---------------------------------------------------------------------------
// Pack fp32 NCHW activations -> bf16 blocked [n][cb=8][p=HW][32ch]
// ---------------------------------------------------------------------------
__global__ void fcos_pack_input(const float* __restrict__ x,
                                __bf16* __restrict__ out,
                                int HW, long long total) {
  long long i = (long long)blockIdx.x * blockDim.x + threadIdx.x;
  if (i >= total) return;
  int c = (int)(i & 31);
  long long t = i >> 5;
  int p = (int)(t % HW);
  long long t2 = t / HW;
  int cb = (int)(t2 & 7);
  int n  = (int)(t2 >> 3);
  out[i] = f2bf(x[((long long)(n * 256 + cb * 32 + c)) * HW + p]);
}

// ---------------------------------------------------------------------------
// Pack conv weights (fp32 OIHW, BN scale folded) -> bf16 A-matrix fragments:
//   pw[head][layer][tap][cb][ocTile][lane][16]
// A 16x32 bf16 layout: lanes 0-15 M=lane hold K {0..7,16..23};
//                      lanes 16-31 M=lane-16 hold K {8..15,24..31}.
// ---------------------------------------------------------------------------
__global__ void fcos_pack_weights(const float* __restrict__ cls_w,
                                  const float* __restrict__ cls_s,
                                  const float* __restrict__ reg_w,
                                  const float* __restrict__ reg_s,
                                  __bf16* __restrict__ pw, long long total) {
  long long i = (long long)blockIdx.x * blockDim.x + threadIdx.x;
  if (i >= total) return;
  long long e = i % 589824;            // 256*256*9 elems per layer
  int tap  = (int)(e >> 16);           // 0..8
  int cb   = (int)((e >> 13) & 7);
  int ocT  = (int)((e >> 9) & 15);
  int lane = (int)((e >> 4) & 31);
  int j    = (int)(e & 15);
  long long rest = i / 589824;
  int layer = (int)(rest & 3);
  int head  = (int)(rest >> 2);
  int ic = cb * 32 + ((lane & 16) ? 8 : 0) + ((j < 8) ? j : j + 8);
  int oc = ocT * 16 + (lane & 15);
  const float* w  = head ? reg_w : cls_w;
  const float* sc = head ? reg_s : cls_s;
  float v = w[(((long long)(layer * 256 + oc) * 256 + ic) * 9) + tap] *
            sc[layer * 256 + oc];
  pw[i] = f2bf(v);
}

// ---------------------------------------------------------------------------
// 3x3 conv (SAME, zero pad) + folded-BN bias + ReLU, bf16 blocked layout,
// fp32 WMMA accumulation. Block: 256 threads = 8 waves, computes
// 256 oc x 64 px; wave w -> ocTiles {2w,2w+1} x pxTiles {0..3} = 8 WMMA/step.
// ---------------------------------------------------------------------------
__global__ __launch_bounds__(256)
void fcos_conv3x3_wmma(const __bf16* __restrict__ actIn,
                       const __bf16* __restrict__ pw,    // layer-packed frags
                       const float*  __restrict__ bias,  // [256]
                       __bf16* __restrict__ actOut,
                       int W, int H, int HW) {
  __shared__ __bf16 sB[2][64 * 32];     // 2 x (64 px x 32 ch) = 2 x 4KB

  const int tid  = threadIdx.x;
  const int wave = tid >> 5;
  const int lane = tid & 31;
  const int pBase = blockIdx.x * 64;    // 64 consecutive row-major pixels
  const int n    = pBase / HW;
  const int pImg = pBase % HW;

  // Staging role: thread handles (pixel = tid>>2, 8 channels = (tid&3)*8), 16B.
  const int sPx = tid >> 2;
  const int sCh = (tid & 3) * 8;
  const int sp  = pImg + sPx;
  const int sh  = sp / W;
  const int sw  = sp % W;
  const __bf16* base0 =
      actIn + (long long)n * 8 * HW * 32 + (long long)sp * 32 + sCh;

  // Per-thread tap validity masks: bit d <-> offset d-1 in that axis.
  const unsigned vy = (sh >= 1 ? 1u : 0u) | 2u | (sh + 1 < H ? 4u : 0u);
  const unsigned vx = (sw >= 1 ? 1u : 0u) | 2u | (sw + 1 < W ? 4u : 0u);

  v8f acc[2][4] = {};                   // [ocTile][pxTile]
  const int ocT0 = wave * 2;
  const int l15  = lane & 15;
  const int chLo = (lane & 16) ? 8 : 0;
  __bf16* sMy = &sB[0][sPx * 32 + sCh];
  const v8bf vzero = {};

  // --- stage step 0: tap (-1,-1), cb 0 ---
  {
    unsigned valid = (vy & 1u) & (vx & 1u);
    const __bf16* gp = base0 + (valid ? (-W - 1) * 32 : 0);
    v8bf v = *(const v8bf*)gp;
    *(v8bf*)sMy = valid ? v : vzero;
  }

  // Incremental staging state describing step sn = s+1 (starts at sn=1:
  // tap 0 = (d3,m3)=(0,0), cb 1). Updated at the bottom of each iteration.
  int cbN = 1, d3N = 0, m3N = 0;
  unsigned validN = (vy & 1u) & (vx & 1u);
  int tapOffN = (-W - 1) * 32;                       // uniform
  const __bf16* gCur = base0 + (long long)HW * 32;   // cb=1 base

  const __bf16* pwPtr = pw + (long long)ocT0 * 512 + lane * 16;

#pragma unroll 2
  for (int s = 0; s < 72; ++s) {        // 9 taps x 8 ic-chunks
    __syncthreads();
    const int cur = s & 1;

    // Branchless staging load for step s+1 (latency hidden under WMMA).
    v8bf vNext = vzero;
    if (s + 1 < 72) {
      const __bf16* gp = gCur + (validN ? tapOffN : 0);
      v8bf v = *(const v8bf*)gp;
      vNext = validN ? v : vzero;
    }
    if (s + 2 < 72)                     // global_prefetch_b8 on next weights
      __builtin_prefetch(pwPtr + 2 * 8192, 0, 0);

    // A fragments: 32B contiguous per lane, pre-packed order.
    v16bf a0 = *(const v16bf*)(pwPtr);
    v16bf a1 = *(const v16bf*)(pwPtr + 512);
    pwPtr += 8192;

    // B fragments from LDS (ds_load_b128 x2 each) + 8 WMMA.
    const __bf16* bbase = &sB[cur][0];
#pragma unroll
    for (int t = 0; t < 4; ++t) {
      const __bf16* pb = bbase + (t * 16 + l15) * 32 + chLo;
      v8bf lo = *(const v8bf*)(pb);
      v8bf hi = *(const v8bf*)(pb + 16);
      v16bf b = __builtin_shufflevector(lo, hi, 0, 1, 2, 3, 4, 5, 6, 7,
                                        8, 9, 10, 11, 12, 13, 14, 15);
      acc[0][t] = __builtin_amdgcn_wmma_f32_16x16x32_bf16(
          false, a0, false, b, (short)0, acc[0][t], false, false);
      acc[1][t] = __builtin_amdgcn_wmma_f32_16x16x32_bf16(
          false, a1, false, b, (short)0, acc[1][t], false, false);
    }

    if (s + 1 < 72)
      *(v8bf*)(sMy + (cur ^ 1) * (64 * 32)) = vNext;

    // Advance staging state to describe step s+2 (uniform branch 1-in-8).
    ++cbN;
    gCur += (long long)HW * 32;
    if (cbN == 8) {
      cbN = 0;
      gCur = base0;
      ++m3N;
      if (m3N == 3) { m3N = 0; ++d3N; }
      validN = ((vy >> d3N) & 1u) & ((vx >> m3N) & 1u);
      tapOffN = ((d3N - 1) * W + (m3N - 1)) * 32;
    }
  }

  // Epilogue: bias + ReLU, pack 8 consecutive oc -> one b128 store.
  // C/D layout: lane 0-15 -> M=r, lane 16-31 -> M=r+8; N = lane&15 (pixel).
  const int mHi = (lane >> 4) * 8;
#pragma unroll
  for (int o = 0; o < 2; ++o) {
    const int ocBase = (ocT0 + o) * 16 + mHi;    // multiple of 8
    const v8f bv = *(const v8f*)(bias + ocBase); // 32B aligned
    const int cbOut = ocBase >> 5;
    const int cOut  = ocBase & 31;
#pragma unroll
    for (int t = 0; t < 4; ++t) {
      const int px = pImg + t * 16 + l15;
      v8bf pkt;
#pragma unroll
      for (int r = 0; r < 8; ++r)
        pkt[r] = (__bf16)fmaxf(acc[o][t][r] + bv[r], 0.0f);  // hw cvt
      *(v8bf*)(actOut + ((long long)(n * 8 + cbOut) * HW + px) * 32 + cOut) = pkt;
    }
  }
}

// ---------------------------------------------------------------------------
// Final 1x1 conv + bias. cls: 80ch raw. reg: ch0 -> centerness (raw),
// ch1..4 -> max(raw*stride, 0). ~0.5% of total FLOPs -> plain fp32 dot.
// Thread = (n, oc, pixel), oc-major so weight rows are wave-uniform.
// ---------------------------------------------------------------------------
__global__ void fcos_head_final(const __bf16* __restrict__ act,
                                const float* __restrict__ fw,
                                const float* __restrict__ fb,
                                float* __restrict__ out0,   // cls or centerness
                                float* __restrict__ out1,   // reg (or unused)
                                int HW, int OC, int isReg, float strideScale,
                                long long total) {
  long long idx = (long long)blockIdx.x * blockDim.x + threadIdx.x;
  if (idx >= total) return;
  int p  = (int)(idx % HW);
  int oc = (int)((idx / HW) % OC);
  int n  = (int)(idx / ((long long)HW * OC));

  float sum = fb[oc];
#pragma unroll 1
  for (int cb = 0; cb < 8; ++cb) {
    const __bf16* b = act + ((long long)(n * 8 + cb) * HW + p) * 32;
    const float* wr = fw + oc * 256 + cb * 32;
#pragma unroll
    for (int c = 0; c < 32; ++c)
      sum = fmaf(bf2f(b[c]), wr[c], sum);
  }

  if (!isReg) {
    out0[((long long)(n * OC) + oc) * HW + p] = sum;
  } else if (oc == 0) {
    out0[(long long)n * HW + p] = sum;                       // centerness
  } else {
    out1[((long long)(n * 4) + (oc - 1)) * HW + p] = fmaxf(sum * strideScale, 0.0f);
  }
}

// ---------------------------------------------------------------------------
// Host orchestration.
// ---------------------------------------------------------------------------
extern "C" void kernel_launch(void* const* d_in, const int* in_sizes, int n_in,
                              void* d_out, int out_size, void* d_ws, size_t ws_size,
                              hipStream_t stream) {
  (void)in_sizes; (void)n_in; (void)out_size; (void)ws_size;

  static const int HWs[5] = {16384, 4096, 1024, 256, 64};
  static const int Wd[5]  = {128, 64, 32, 16, 8};
  static const float STR[5] = {8.f, 16.f, 32.f, 64.f, 128.f};
  const int NB = 4;

  const float* fpn[5];
  for (int l = 0; l < 5; ++l) fpn[l] = (const float*)d_in[l];
  const float* cls_conv_w = (const float*)d_in[5];
  const float* cls_bn_s   = (const float*)d_in[6];
  const float* cls_bn_b   = (const float*)d_in[7];
  const float* cls_fw     = (const float*)d_in[8];
  const float* cls_fb     = (const float*)d_in[9];
  const float* reg_conv_w = (const float*)d_in[10];
  const float* reg_bn_s   = (const float*)d_in[11];
  const float* reg_bn_b   = (const float*)d_in[12];
  const float* reg_fw     = (const float*)d_in[13];
  const float* reg_fb     = (const float*)d_in[14];
  float* out = (float*)d_out;

  // d_out layout: cls0..4, reg0..4, ctr0..4 (fp32, NCHW each)
  long long clsOff[5], regOff[5], ctrOff[5], off = 0;
  for (int l = 0; l < 5; ++l) { clsOff[l] = off; off += 320LL * HWs[l]; }
  for (int l = 0; l < 5; ++l) { regOff[l] = off; off +=  16LL * HWs[l]; }
  for (int l = 0; l < 5; ++l) { ctrOff[l] = off; off +=   4LL * HWs[l]; }

  // ws layout
  const long long PW_ELEMS  = 2LL * 4 * 589824;          // 4,718,592 bf16
  const long long PW_BYTES  = PW_ELEMS * 2;              // ~9.4 MB
  const long long BUF_BYTES = 4LL * 16384 * 256 * 2;     // 33.5 MB (level 0)
  __bf16* pw    = (__bf16*)d_ws;
  __bf16* bufIn = (__bf16*)((char*)d_ws + PW_BYTES);
  __bf16* bufA  = (__bf16*)((char*)d_ws + PW_BYTES + BUF_BYTES);
  __bf16* bufB  = (__bf16*)((char*)d_ws + PW_BYTES + 2 * BUF_BYTES);

  // 1) pack weights (both heads, BN scale folded)
  fcos_pack_weights<<<(unsigned)(PW_ELEMS / 256), 256, 0, stream>>>(
      cls_conv_w, cls_bn_s, reg_conv_w, reg_bn_s, pw, PW_ELEMS);

  for (int l = 0; l < 5; ++l) {
    const int HW = HWs[l], W = Wd[l], H = HW / W;
    const long long packTotal = (long long)NB * 256 * HW;
    fcos_pack_input<<<(unsigned)(packTotal / 256), 256, 0, stream>>>(
        fpn[l], bufIn, HW, packTotal);

    const unsigned convBlocks = (unsigned)((NB * HW) / 64);
    for (int head = 0; head < 2; ++head) {
      const __bf16* pwh  = pw + (long long)head * 4 * 589824;
      const float*  bnb  = head ? reg_bn_b : cls_bn_b;

      fcos_conv3x3_wmma<<<convBlocks, 256, 0, stream>>>(
          bufIn, pwh + 0LL * 589824, bnb + 0,   bufA, W, H, HW);
      fcos_conv3x3_wmma<<<convBlocks, 256, 0, stream>>>(
          bufA,  pwh + 1LL * 589824, bnb + 256, bufB, W, H, HW);
      fcos_conv3x3_wmma<<<convBlocks, 256, 0, stream>>>(
          bufB,  pwh + 2LL * 589824, bnb + 512, bufA, W, H, HW);
      fcos_conv3x3_wmma<<<convBlocks, 256, 0, stream>>>(
          bufA,  pwh + 3LL * 589824, bnb + 768, bufB, W, H, HW);

      if (head == 0) {
        const long long tot = (long long)NB * 80 * HW;
        fcos_head_final<<<(unsigned)((tot + 255) / 256), 256, 0, stream>>>(
            bufB, cls_fw, cls_fb, out + clsOff[l], nullptr,
            HW, 80, 0, 1.0f, tot);
      } else {
        const long long tot = (long long)NB * 5 * HW;
        fcos_head_final<<<(unsigned)((tot + 255) / 256), 256, 0, stream>>>(
            bufB, reg_fw, reg_fb, out + ctrOff[l], out + regOff[l],
            HW, 5, 1, STR[l], tot);
      }
    }
  }
}
